// SepLinProjSum_18021682774670
// MI455X (gfx1250) — matrix operand order repaired and checked by reading, not compiled, not run on hardware
//
#include <hip/hip_runtime.h>

typedef __attribute__((ext_vector_type(16))) __bf16 v16bf;
typedef __attribute__((ext_vector_type(2)))  __bf16 v2bf;
typedef __attribute__((ext_vector_type(8)))  float  v8f;
typedef __attribute__((ext_vector_type(4)))  float  v4f;
typedef __attribute__((ext_vector_type(4)))  unsigned int v4u;
typedef __attribute__((ext_vector_type(4)))  int    v4i;

#define TOKN   256
#define NSTEP  18            // ceil(568/32) -> K padded to 576
#define MT     64            // rows per block
#define AST    584           // LDS A row stride (bf16 units) = 1168 B = 73*16B (conflict-free)
#define WST    40            // LDS W per-col stride (bf16 units) = 80 B = 5*16B (conflict-free)
#define WSLAB  (TOKN * WST)  // 10240 ushorts = 20480 B per slab

// Async global->LDS path (gfx1250 GLOBAL_LOAD_ASYNC_TO_LDS_B128, ASYNCcnt-tracked)
#if defined(__has_builtin)
#if __has_builtin(__builtin_amdgcn_global_load_async_to_lds_b128) && \
    __has_builtin(__builtin_amdgcn_s_wait_asynccnt)
#define USE_ASYNC_LDS 1
#endif
#if __has_builtin(__builtin_amdgcn_cvt_pk_bf16_f32)
#define USE_PK_CVT 1
#endif
#endif

__device__ __forceinline__ unsigned short f2bf(float f) {
  union { float f; unsigned int u; } c; c.f = f;
  unsigned int u = c.u;
  u += 0x7FFFu + ((u >> 16) & 1u);   // round-to-nearest-even
  return (unsigned short)(u >> 16);
}

__device__ __forceinline__ unsigned int pk2bf(float x, float y) {
#ifdef USE_PK_CVT
  v2bf p = __builtin_amdgcn_cvt_pk_bf16_f32(x, y);   // v_cvt_pk_bf16_f32 (RNE)
  union { v2bf v; unsigned int u; } c; c.v = p;
  return c.u;
#else
  return (unsigned int)f2bf(x) | ((unsigned int)f2bf(y) << 16);
#endif
}

#ifdef USE_ASYNC_LDS
typedef __attribute__((address_space(1))) v4i g_v4i;   // global int4
typedef __attribute__((address_space(3))) v4i l_v4i;   // LDS int4

__device__ __forceinline__ g_v4i* glb_cast(const void* p) {
  return (g_v4i*)(unsigned long long)p;                // global generic addr == AS1 addr
}
// ISA 10.2: generic LDS address -> LDS offset is addr[31:0] truncation.
__device__ __forceinline__ l_v4i* lds_cast(void* p) {
  return (l_v4i*)(unsigned int)(unsigned long long)p;
}
#endif

union FragB { v16bf v; v4u q[2]; };

// Pack app_W[513,256] ++ st_W[55,256] (fp32) into bf16 slabs in workspace.
// wc layout: [NSTEP][col t][WST k]  (k 0..31 valid, global K = s*32+k; rest zero)
__global__ void pack_wc(const float* __restrict__ appW, const float* __restrict__ stW,
                        unsigned short* __restrict__ wc) {
  int gid = blockIdx.x * blockDim.x + threadIdx.x;   // NSTEP*256 threads
  int s = gid >> 8;
  int t = gid & 255;
  unsigned short* dst = wc + (size_t)(s * TOKN + t) * WST;
  #pragma unroll
  for (int kk = 0; kk < WST; ++kk) {
    int k = s * 32 + kk;
    float v = 0.0f;
    if (kk < 32) {
      if (k < 513)      v = appW[(size_t)k * TOKN + t];          // emb(512) + vis row
      else if (k < 568) v = stW[(size_t)(k - 513) * TOKN + t];   // bbox+kpts rows
    }
    dst[kk] = f2bf(v);
  }
}

__global__ __launch_bounds__(256)
void fused_tok(const float* __restrict__ emb, const float* __restrict__ vis,
               const float* __restrict__ bbox, const float* __restrict__ kpt,
               const int* __restrict__ mask,
               const float* __restrict__ appb, const float* __restrict__ stb,
               const unsigned short* __restrict__ wc,
               float* __restrict__ out) {
  extern __shared__ unsigned short smem[];
  unsigned short* Asm = smem;                  // MT*AST bf16
  unsigned short* Wsm = smem + MT * AST;       // 2*WSLAB bf16 (double buffer)

  const int tid  = threadIdx.x;
  const int lane = tid & 31;
  const int wave = tid >> 5;
  const long rowBase = (long)blockIdx.x * MT;

  // ---- Kick off W slab 0 staging first (overlaps the A-tile VALU conversion) ----
#ifdef USE_ASYNC_LDS
  {
    #pragma unroll
    for (int i = 0; i < 5; ++i) {
      const int off = (tid + i * 256) * 8;     // 16 B chunks, ushort units
      __builtin_amdgcn_global_load_async_to_lds_b128(glb_cast(wc + off),
                                                     lds_cast(Wsm + off), 0, 0);
    }
  }
#endif

  // ---- Stage A tile: embeddings (64 rows x 512) fp32 -> bf16 in LDS ----
  {
    const v4f* ev = reinterpret_cast<const v4f*>(emb) + rowBase * 128;  // 128 chunks per row
    #pragma unroll 4
    for (int i = tid; i < MT * 128; i += 256) {
      v4f v = __builtin_nontemporal_load(&ev[i]);     // streamed once; don't pollute caches
      int r = i >> 7, c = (i & 127) << 2;
      unsigned int* d = reinterpret_cast<unsigned int*>(&Asm[r * AST + c]);  // 8B aligned
      d[0] = pk2bf(v.x, v.y);
      d[1] = pk2bf(v.z, v.w);
    }
  }
  // ---- Stage A tail: k = 512..575 -> [vis | bbox(4) | kpts(51) | zero pad] ----
  {
    int r  = tid >> 2;                 // 4 threads per row
    int k0 = 512 + ((tid & 3) << 4);   // 16 k's each
    long row = rowBase + r;
    #pragma unroll
    for (int j = 0; j < 16; ++j) {
      int k = k0 + j;
      float v = 0.0f;
      if (k == 512)      v = vis[row];
      else if (k < 517)  v = bbox[row * 4 + (k - 513)];
      else if (k < 568)  v = kpt[row * 51 + (k - 517)];
      Asm[r * AST + k] = f2bf(v);
    }
  }
#ifndef USE_ASYNC_LDS
  // ---- Stage W slab 0 (fallback: through VGPRs) ----
  {
    const v4u* s0 = reinterpret_cast<const v4u*>(wc);
    v4u* d0 = reinterpret_cast<v4u*>(Wsm);
    #pragma unroll
    for (int i = 0; i < 5; ++i) d0[tid + i * 256] = s0[tid + i * 256];
  }
#else
  __builtin_amdgcn_s_wait_asynccnt(0);
#endif
  __syncthreads();

  // wave -> (M subtile, N half); each wave: 16 rows x 128 cols = 8 accum tiles
  const int mt = wave & 3;
  const int nh = wave >> 2;
  const int halfsel = lane >> 4;                                    // 0 | 1
  const int aoff0 = (mt * 16 + (lane & 15)) * AST + (halfsel << 3); // + s*32 per step
  const int koff  = halfsel << 3;                                   // B frag K offset 0|8
  const int tloc  = (nh * 128 + (lane & 15)) * WST + koff;          // B frag base (nt += 16*WST)

  v8f acc[8];
  #pragma unroll
  for (int i = 0; i < 8; ++i)
    #pragma unroll
    for (int e = 0; e < 8; ++e) acc[i][e] = 0.0f;

  // A fragment for step 0 (A tile is stable in LDS; safe to prefetch next steps early)
  FragB a_cur;
  {
    const v4u* ap = reinterpret_cast<const v4u*>(&Asm[aoff0]);
    a_cur.q[0] = ap[0];
    a_cur.q[1] = ap[2];
  }

  for (int s = 0; s < NSTEP; ++s) {
    const int buf = s & 1;
    const bool more = (s + 1 < NSTEP);

    // Stage next W slab into the other LDS buffer (overlaps WMMAs below)
#ifdef USE_ASYNC_LDS
    if (more) {
      const unsigned short* src = wc + (size_t)(s + 1) * WSLAB;
      unsigned short* dst = &Wsm[(buf ^ 1) * WSLAB];
      #pragma unroll
      for (int i = 0; i < 5; ++i) {
        const int off = (tid + i * 256) * 8;
        __builtin_amdgcn_global_load_async_to_lds_b128(glb_cast(src + off),
                                                       lds_cast(dst + off), 0, 0);
      }
    }
#else
    v4u pf[5];
    if (more) {
      const v4u* src = reinterpret_cast<const v4u*>(wc + (size_t)(s + 1) * WSLAB);
      #pragma unroll
      for (int i = 0; i < 5; ++i) pf[i] = src[tid + i * 256];
    }
#endif

    // Prefetch next step's A fragment early
    FragB a_nxt;
    if (more) {
      const v4u* ap = reinterpret_cast<const v4u*>(&Asm[aoff0 + (s + 1) * 32]);
      a_nxt.q[0] = ap[0];
      a_nxt.q[1] = ap[2];
    }

    // Ping-pong B fragments so ds_load latency hides behind the previous WMMA
    const unsigned short* wb = &Wsm[buf * WSLAB];
    FragB bcur, bnxt;
    {
      const v4u* bp = reinterpret_cast<const v4u*>(&wb[tloc]);
      bcur.q[0] = bp[0];
      bcur.q[1] = bp[2];
    }
    #pragma unroll
    for (int nt = 0; nt < 8; ++nt) {
      if (nt < 7) {
        const v4u* bp = reinterpret_cast<const v4u*>(&wb[tloc + (nt + 1) * 16 * WST]);
        bnxt.q[0] = bp[0];
        bnxt.q[1] = bp[2];
      }
      acc[nt] = __builtin_amdgcn_wmma_f32_16x16x32_bf16(
          /*neg_a=*/false, a_cur.v, /*neg_b=*/false, bcur.v,
          /*c_mod=*/(short)0, acc[nt], /*reuse_a=*/false, /*reuse_b=*/false);
      bcur = bnxt;
    }

#ifdef USE_ASYNC_LDS
    if (more) __builtin_amdgcn_s_wait_asynccnt(0);
#else
    if (more) {
      v4u* dst = reinterpret_cast<v4u*>(&Wsm[(buf ^ 1) * WSLAB]);
      #pragma unroll
      for (int i = 0; i < 5; ++i) dst[tid + i * 256] = pf[i];
    }
#endif
    a_cur = a_nxt;
    __syncthreads();
  }

  // ---- Epilogue: bias + row mask + store ----
  const int mrow0 = mt * 16 + (halfsel << 3);   // C tile: VGPR j -> row mrow0+j
  int mok[8];
  #pragma unroll
  for (int j = 0; j < 8; ++j) mok[j] = mask[rowBase + mrow0 + j];

  #pragma unroll
  for (int nt = 0; nt < 8; ++nt) {
    const int t = nh * 128 + nt * 16 + (lane & 15);
    const float bias = appb[t] + stb[t];
    #pragma unroll
    for (int j = 0; j < 8; ++j) {
      const long r = rowBase + mrow0 + j;
      const float v = mok[j] ? (acc[nt][j] + bias) : 0.0f;
      __builtin_nontemporal_store(v, &out[r * TOKN + t]);
    }
  }
}

extern "C" void kernel_launch(void* const* d_in, const int* in_sizes, int n_in,
                              void* d_out, int out_size, void* d_ws, size_t ws_size,
                              hipStream_t stream) {
  const float* emb  = (const float*)d_in[0];   // [B,N,512]
  const float* vis  = (const float*)d_in[1];   // [B,N,1]
  const float* bbox = (const float*)d_in[2];   // [B,N,4]
  const float* kpt  = (const float*)d_in[3];   // [B,N,17,3]
  const int*   mask = (const int*)d_in[4];     // [B,N]
  const float* appW = (const float*)d_in[5];   // [513,256]
  const float* appb = (const float*)d_in[6];   // [256]
  const float* stW  = (const float*)d_in[7];   // [55,256]
  const float* stb  = (const float*)d_in[8];   // [256]
  float* out = (float*)d_out;
  unsigned short* wc = (unsigned short*)d_ws;  // needs NSTEP*WSLAB*2 = 368,640 B

  // 1) pack weights to bf16 slabs (tiny; L2-resident)
  pack_wc<<<NSTEP, 256, 0, stream>>>(appW, stW, wc);

  // 2) fused GEMM + bias + mask
  const int Mtot = in_sizes[4];                // B*N = 262144 rows
  const int grid = Mtot / MT;                  // 4096 blocks
  const size_t smemBytes = (size_t)(MT * AST + 2 * WSLAB) * sizeof(unsigned short); // ~113 KB
  fused_tok<<<grid, 256, smemBytes, stream>>>(emb, vis, bbox, kpt, mask, appb, stb, wc, out);

  (void)n_in; (void)out_size; (void)ws_size;
}
